// SpectralConv2d_20306605375634
// MI455X (gfx1250) — matrix operand order repaired and checked by reading, not compiled, not run on hardware
//
#include <hip/hip_runtime.h>
#include <hip/hip_bf16.h>

typedef __bf16 bf16_t;
typedef __attribute__((ext_vector_type(16))) __bf16 v16bf;
typedef __attribute__((ext_vector_type(8)))  float  v8f;
typedef __attribute__((ext_vector_type(4)))  int    v4i;

#define KDIM     2048              // 64*32
#define NDIM     2048
#define MROWS    16384             // 4*64*64
#define XELEMS   33554432          // 4*64^3*32
#define BSTRIDE  8388608           // 64^3 * 32  (batch stride in elements)
#define PI_F     3.14159265358979323846f

#if __has_builtin(__builtin_amdgcn_global_load_async_to_lds_b128) && \
    __has_builtin(__builtin_amdgcn_s_wait_asynccnt)
#define USE_ASYNC_LDS 1
#else
#define USE_ASYNC_LDS 0
#endif

__device__ __forceinline__ bf16_t to_bf16(float f) {
    unsigned int u = __builtin_bit_cast(unsigned int, f);
    unsigned int r = (u + 0x7FFFu + ((u >> 16) & 1u)) >> 16;
    return __builtin_bit_cast(bf16_t, (unsigned short)r);
}

__device__ __forceinline__ unsigned int pack_bf16_2(float a, float b) {
    unsigned int ua = __builtin_bit_cast(unsigned int, a);
    unsigned int ub = __builtin_bit_cast(unsigned int, b);
    unsigned int ra = (ua + 0x7FFFu + ((ua >> 16) & 1u)) >> 16;
    unsigned int rb = (ub + 0x7FFFu + ((ub >> 16) & 1u)) >> 16;
    return (ra & 0xFFFFu) | (rb << 16);
}

#if USE_ASYNC_LDS
__device__ __forceinline__ void async_cp_b128(const void* g, void* l) {
    __attribute__((address_space(1))) v4i* gp =
        (__attribute__((address_space(1))) v4i*)(v4i*)(void*)(size_t)(const char*)g;
    __attribute__((address_space(3))) v4i* lp =
        (__attribute__((address_space(3))) v4i*)(v4i*)l;
    __builtin_amdgcn_global_load_async_to_lds_b128(gp, lp, 0, 0);
}
#endif

__device__ __forceinline__ v8f wmma_bf16(v16bf a, v16bf b, v8f c) {
    return __builtin_amdgcn_wmma_f32_16x16x32_bf16(false, a, false, b, (short)0, c, false, false);
}

// A-fragment (16x32 bf16): lane = kh*16 + m; elems 0..15 -> K=kh*16+0..15,
// 16 contiguous bf16 at row[kh*16].
__device__ __forceinline__ v16bf load_frag_a(const bf16_t* base, int ld) {
    int lane = threadIdx.x & 31;
    int m = lane & 15, kh = lane >> 4;
    const bf16_t* row = base + m * ld;
    v16bf a;
#pragma unroll
    for (int e = 0; e < 16; ++e) a[e] = row[kh * 16 + e];
    return a;
}

// B-fragment from N-major tile Bt[n][k]: lane = kh*16 + n;
// VGPR r = K=kh*16+2r,2r+1  ==  16 contiguous bf16 at Bt[n][kh*16].
__device__ __forceinline__ v16bf load_frag_b_t(const bf16_t* base, int ld) {
    int lane = threadIdx.x & 31;
    int n = lane & 15, kh = lane >> 4;
    const bf16_t* p = base + n * ld + kh * 16;
    v16bf b;
#pragma unroll
    for (int e = 0; e < 16; ++e) b[e] = p[e];
    return b;
}

// ---------------------------------------------------------------------------
// Stage -1: convert x (f32) -> bf16 once; removes all cvt VALU from GEMM loops
// ---------------------------------------------------------------------------
__global__ __launch_bounds__(256) void convert_bf16(const float* __restrict__ x,
                                                    bf16_t* __restrict__ xb) {
    size_t idx = ((size_t)blockIdx.x * 256 + threadIdx.x) * 16;
    float4 v0 = *(const float4*)(x + idx);
    float4 v1 = *(const float4*)(x + idx + 4);
    float4 v2 = *(const float4*)(x + idx + 8);
    float4 v3 = *(const float4*)(x + idx + 12);
    uint4 q0, q1;
    q0.x = pack_bf16_2(v0.x, v0.y); q0.y = pack_bf16_2(v0.z, v0.w);
    q0.z = pack_bf16_2(v1.x, v1.y); q0.w = pack_bf16_2(v1.z, v1.w);
    q1.x = pack_bf16_2(v2.x, v2.y); q1.y = pack_bf16_2(v2.z, v2.w);
    q1.z = pack_bf16_2(v3.x, v3.y); q1.w = pack_bf16_2(v3.z, v3.w);
    *(uint4*)(xb + idx)     = q0;
    *(uint4*)(xb + idx + 8) = q1;
}

// ---------------------------------------------------------------------------
// Stage 0: combined per-axis operator, stored N-MAJOR:  Tt[n][k], n=(s',o), k=(s,i)
// Tt[(s',o)][(s,i)] = sum_m c_m^2 cos(pi(2s+1)m/128) cos(pi(2s'+1)m/128) * w[i,o,m]
// ---------------------------------------------------------------------------
__global__ __launch_bounds__(256) void build_T(const float* __restrict__ w,
                                               bf16_t* __restrict__ Tt) {
    int idx = blockIdx.x * 256 + threadIdx.x;   // n*2048 + k
    int k = idx & (KDIM - 1);
    int n = idx >> 11;
    int s  = k >> 5, i = k & 31;
    int sp = n >> 5, o = n & 31;
    const float* wv = w + (i * 32 + o) * 16;
    float acc = 0.0f;
#pragma unroll
    for (int m = 0; m < 16; ++m) {
        float cm2 = (m == 0) ? (1.0f / 64.0f) : (1.0f / 32.0f);
        float ca = __cosf((2 * s + 1) * m * (PI_F / 128.0f));
        float cb = __cosf((2 * sp + 1) * m * (PI_F / 128.0f));
        acc += cm2 * ca * cb * wv[m];
    }
    Tt[idx] = to_bf16(acc);
}

// ---------------------------------------------------------------------------
// Stage 1: per-axis GEMM, 128x128 block, 8 waves, wave tile 32x64 (2x4 WMMA).
// Double-buffered LDS; both tiles staged with async Global->LDS 128-bit DMA
// (ASYNCcnt) when available, else 128-bit VGPR copies.
// ---------------------------------------------------------------------------
__global__ __launch_bounds__(256) void axis_gemm(const bf16_t* __restrict__ xb,
                                                 const bf16_t* __restrict__ Tt,
                                                 float* __restrict__ out,
                                                 int sS, int c1S, int c2S, int accum) {
    __shared__ bf16_t As[2][128][40];   // rows x K(32), padded
    __shared__ bf16_t Bt[2][128][40];   // n x K(32), padded (N-major)

    const int t    = threadIdx.x;
    const int row0 = blockIdx.x * 128;
    const int n0   = blockIdx.y * 128;

    // A staging: thread -> (row, 16-elem half); fixed across K loop
    const int aRow = t >> 1;
    const int aCol = (t & 1) * 16;
    int gr = row0 + aRow;
    int ab = gr >> 12, ac1 = (gr >> 6) & 63, ac2 = gr & 63;
    const bf16_t* aSrc = xb + (size_t)ab * BSTRIDE + (size_t)ac1 * c1S +
                         (size_t)ac2 * c2S + aCol;

    // B staging: thread -> (n row, 16-elem half of the 32-wide K chunk)
    const int bN = t >> 1;
    const int bK = (t & 1) * 16;
    const bf16_t* bSrc = Tt + (size_t)(n0 + bN) * KDIM + bK;

    const int w     = t >> 5;
    const int waveM = (w & 3) * 32;     // 0,32,64,96
    const int waveN = (w >> 2) * 64;    // 0,64

    const v8f vzero = {0.f, 0.f, 0.f, 0.f, 0.f, 0.f, 0.f, 0.f};
    v8f acc[2][4];
#pragma unroll
    for (int mt = 0; mt < 2; ++mt)
#pragma unroll
        for (int nt = 0; nt < 4; ++nt) acc[mt][nt] = vzero;

    // ---- prologue: stage K-chunk 0 into buffer 0
#if USE_ASYNC_LDS
    async_cp_b128(aSrc,     &As[0][aRow][aCol]);
    async_cp_b128(aSrc + 8, &As[0][aRow][aCol + 8]);
    async_cp_b128(bSrc,     &Bt[0][bN][bK]);
    async_cp_b128(bSrc + 8, &Bt[0][bN][bK + 8]);
    __builtin_amdgcn_s_wait_asynccnt(0);
#else
    *(uint4*)&As[0][aRow][aCol]     = *(const uint4*)(aSrc);
    *(uint4*)&As[0][aRow][aCol + 8] = *(const uint4*)(aSrc + 8);
    *(uint4*)&Bt[0][bN][bK]         = *(const uint4*)(bSrc);
    *(uint4*)&Bt[0][bN][bK + 8]     = *(const uint4*)(bSrc + 8);
#endif
    __syncthreads();

    int buf = 0;
    for (int ks = 0; ks < 64; ++ks) {
        // ---- stage next chunk into the other buffer (overlaps WMMA below)
        if (ks + 1 < 64) {
            const bf16_t* src = aSrc + (size_t)(ks + 1) * sS;
            const bf16_t* sb  = bSrc + (size_t)(ks + 1) * 32;
            __builtin_prefetch(aSrc + (size_t)(ks + 2) * sS, 0, 1);  // global_prefetch_b8
#if USE_ASYNC_LDS
            async_cp_b128(src,     &As[buf ^ 1][aRow][aCol]);
            async_cp_b128(src + 8, &As[buf ^ 1][aRow][aCol + 8]);
            async_cp_b128(sb,      &Bt[buf ^ 1][bN][bK]);
            async_cp_b128(sb + 8,  &Bt[buf ^ 1][bN][bK + 8]);
#else
            *(uint4*)&As[buf ^ 1][aRow][aCol]     = *(const uint4*)(src);
            *(uint4*)&As[buf ^ 1][aRow][aCol + 8] = *(const uint4*)(src + 8);
            *(uint4*)&Bt[buf ^ 1][bN][bK]         = *(const uint4*)(sb);
            *(uint4*)&Bt[buf ^ 1][bN][bK + 8]     = *(const uint4*)(sb + 8);
#endif
        }

        // ---- compute current chunk: 2x4 WMMA tiles
        v16bf af[2], bfr[4];
        af[0] = load_frag_a(&As[buf][waveM][0], 40);
        af[1] = load_frag_a(&As[buf][waveM + 16][0], 40);
#pragma unroll
        for (int nt = 0; nt < 4; ++nt)
            bfr[nt] = load_frag_b_t(&Bt[buf][waveN + nt * 16][0], 40);
#pragma unroll
        for (int mt = 0; mt < 2; ++mt)
#pragma unroll
            for (int nt = 0; nt < 4; ++nt)
                acc[mt][nt] = wmma_bf16(af[mt], bfr[nt], acc[mt][nt]);

#if USE_ASYNC_LDS
        __builtin_amdgcn_s_wait_asynccnt(0);
#endif
        __syncthreads();
        buf ^= 1;
    }

    // ---- epilogue: D layout lane = mh*16 + n; VGPR r -> M = mh*8 + r
    const int lane = t & 31;
    const int nloc = lane & 15, mh = lane >> 4;
#pragma unroll
    for (int mt = 0; mt < 2; ++mt) {
#pragma unroll
        for (int nt = 0; nt < 4; ++nt) {
            int ng = n0 + waveN + nt * 16 + nloc;
            int sp = ng >> 5, o = ng & 31;
#pragma unroll
            for (int r = 0; r < 8; ++r) {
                int m  = waveM + mt * 16 + mh * 8 + r;
                int g  = row0 + m;
                int bb = g >> 12, c1 = (g >> 6) & 63, c2 = g & 63;
                size_t addr = (size_t)bb * BSTRIDE + (size_t)c1 * c1S +
                              (size_t)c2 * c2S + (size_t)sp * sS + o;
                if (accum) out[addr] += acc[mt][nt][r];
                else       out[addr]  = acc[mt][nt][r];
            }
        }
    }
}

// ---------------------------------------------------------------------------
// Stage 2: in-place pointwise MLP on h (= d_out):  relu(h@W0+b0)@W1+b1
// ---------------------------------------------------------------------------
__global__ __launch_bounds__(256) void mlp_kernel(float* __restrict__ h,
                                                  const float* __restrict__ w0,
                                                  const float* __restrict__ b0,
                                                  const float* __restrict__ w1,
                                                  const float* __restrict__ b1) {
    __shared__ bf16_t Xs[128][40];
    __shared__ bf16_t W0t[64][40];      // N-major: [n][k], k=0..31
    __shared__ bf16_t W1t[32][72];      // N-major: [n][k], k=0..63
    __shared__ bf16_t H1s[128][72];

    const int t = threadIdx.x;
    const size_t base = (size_t)blockIdx.x * 128 * 32;

    {   // W0 (32x64, k-major in global) -> W0t[n][k]
        int idx = t * 8, r = idx >> 6, c = idx & 63;
#pragma unroll
        for (int j = 0; j < 8; ++j) W0t[c + j][r] = to_bf16(w0[idx + j]);
    }
    {   // W1 (64x32) -> W1t[n][k]
        int idx = t * 8, r = idx >> 5, c = idx & 31;
#pragma unroll
        for (int j = 0; j < 8; ++j) W1t[c + j][r] = to_bf16(w1[idx + j]);
    }
    {   // X: 128 rows x 32 ch, packed stores
        int r = t >> 1, c = (t & 1) * 16;
        const float* src = h + base + (size_t)r * 32 + c;
        float4 v0 = *(const float4*)(src + 0);
        float4 v1 = *(const float4*)(src + 4);
        float4 v2 = *(const float4*)(src + 8);
        float4 v3 = *(const float4*)(src + 12);
        uint4 q0, q1;
        q0.x = pack_bf16_2(v0.x, v0.y); q0.y = pack_bf16_2(v0.z, v0.w);
        q0.z = pack_bf16_2(v1.x, v1.y); q0.w = pack_bf16_2(v1.z, v1.w);
        q1.x = pack_bf16_2(v2.x, v2.y); q1.y = pack_bf16_2(v2.z, v2.w);
        q1.z = pack_bf16_2(v3.x, v3.y); q1.w = pack_bf16_2(v3.z, v3.w);
        *(uint4*)&Xs[r][c]     = q0;
        *(uint4*)&Xs[r][c + 8] = q1;
    }
    __syncthreads();

    const int w = t >> 5, rowOff = w * 16;
    const int lane = t & 31, nloc = lane & 15, mh = lane >> 4;
    const v8f vzero = {0.f, 0.f, 0.f, 0.f, 0.f, 0.f, 0.f, 0.f};

    // layer 1: (16x32) @ (32x64) + b0, ReLU -> H1s
    v16bf a = load_frag_a(&Xs[rowOff][0], 40);
#pragma unroll
    for (int nt = 0; nt < 4; ++nt) {
        v16bf bfg = load_frag_b_t(&W0t[nt * 16][0], 40);
        v8f c = wmma_bf16(a, bfg, vzero);
        float bias = b0[nt * 16 + nloc];
#pragma unroll
        for (int r = 0; r < 8; ++r) {
            float v = c[r] + bias;
            v = v > 0.0f ? v : 0.0f;
            H1s[rowOff + mh * 8 + r][nt * 16 + nloc] = to_bf16(v);
        }
    }
    __syncthreads();

    // layer 2: (16x64) @ (64x32) + b1
    v8f acc2[2] = {vzero, vzero};
#pragma unroll
    for (int kc = 0; kc < 2; ++kc) {
        v16bf a2 = load_frag_a(&H1s[rowOff][kc * 32], 72);
#pragma unroll
        for (int nt = 0; nt < 2; ++nt) {
            v16bf b2 = load_frag_b_t(&W1t[nt * 16][kc * 32], 72);
            acc2[nt] = wmma_bf16(a2, b2, acc2[nt]);
        }
    }
#pragma unroll
    for (int nt = 0; nt < 2; ++nt) {
        float bias = b1[nt * 16 + nloc];
#pragma unroll
        for (int r = 0; r < 8; ++r) {
            size_t row = (size_t)blockIdx.x * 128 + rowOff + mh * 8 + r;
            h[row * 32 + nt * 16 + nloc] = acc2[nt][r] + bias;
        }
    }
}

extern "C" void kernel_launch(void* const* d_in, const int* in_sizes, int n_in,
                              void* d_out, int out_size, void* d_ws, size_t ws_size,
                              hipStream_t stream) {
    const float* x    = (const float*)d_in[0];
    const float* fw_x = (const float*)d_in[1];
    const float* fw_y = (const float*)d_in[2];
    const float* fw_z = (const float*)d_in[3];
    const float* w0   = (const float*)d_in[4];
    const float* b0   = (const float*)d_in[5];
    const float* w1   = (const float*)d_in[6];
    const float* b1   = (const float*)d_in[7];
    float* out = (float*)d_out;

    bf16_t* Tz = (bf16_t*)d_ws;
    bf16_t* Ty = Tz + (size_t)KDIM * NDIM;
    bf16_t* Tx = Ty + (size_t)KDIM * NDIM;
    bf16_t* xb = Tx + (size_t)KDIM * NDIM;   // 64 MB bf16 copy of x

    convert_bf16<<<XELEMS / (256 * 16), 256, 0, stream>>>(x, xb);

    const int tBlocks = (KDIM * NDIM) / 256;
    build_T<<<tBlocks, 256, 0, stream>>>(fw_z, Tz);
    build_T<<<tBlocks, 256, 0, stream>>>(fw_y, Ty);
    build_T<<<tBlocks, 256, 0, stream>>>(fw_x, Tx);

    dim3 grid(MROWS / 128, NDIM / 128);
    // z-axis: rows (b,X,Y); s stride = 32 (Z).  Writes h.
    axis_gemm<<<grid, 256, 0, stream>>>(xb, Tz, out, 32, 131072, 2048, 0);
    // y-axis: rows (b,X,Z); s stride = 2048 (Y).  Accumulates.
    axis_gemm<<<grid, 256, 0, stream>>>(xb, Ty, out, 2048, 131072, 32, 1);
    // x-axis: rows (b,Y,Z); s stride = 131072 (X).  Accumulates.
    axis_gemm<<<grid, 256, 0, stream>>>(xb, Tx, out, 131072, 2048, 32, 1);

    // pointwise MLP, in place on d_out
    mlp_kernel<<<(4 * 64 * 64 * 64 * 32) / (128 * 32), 256, 0, stream>>>(out, w0, b0, w1, b1);
}
